// PointNet_51582557225105
// MI455X (gfx1250) — matrix operand order repaired and checked by reading, not compiled, not run on hardware
//
#include <hip/hip_runtime.h>

// ---------------------------------------------------------------------------
// Problem constants (match reference setup_inputs)
// ---------------------------------------------------------------------------
static constexpr int N0   = 8192;     // stage-0 nodes
static constexpr int DIN  = 16;       // input feature dim
static constexpr int E0   = 163840;   // stage-0 edges
static constexpr int DH   = 64;       // hidden dim
static constexpr int NL   = 3;        // gnn layers
static constexpr int M0   = 2048;     // stage-0 centroids (N0/4)
static constexpr int M1   = 512;      // stage-1 centroids (M0/4)
static constexpr int TOPK = 20;
static constexpr float RADIUS2 = 0.2f * 0.2f;
static constexpr float E_CONST = 2.718281828459045f;

typedef __attribute__((ext_vector_type(16))) _Float16 v16h;
typedef __attribute__((ext_vector_type(8)))  float    v8f;

// ---------------------------------------------------------------------------
// LDS-staged, double-buffered WMMA GEMM:  C = [relu](A x B + bias)
//   A(i,k) = A[i*ars + k*acs]   (s^T is ars=1, acs=m)
//   B(k,j) = B[k*brs + j*bcs]
// 256 threads = 8 waves, tiled 4(M) x 2(N); each wave does 32x32 via 2x2
// 16x16x32 f16 WMMA fragments.  Workgroup tile: 128(M) x 64(N), K-step 32.
//
// Pipeline: while tile i is multiplied out of LDS buffer i&1, the global
// loads for tile i+1 are already in flight into registers (one barrier per
// K-step).  A is stored in LDS with the two middle k-groups of 8 swapped
// ("perm") so a lane's 16 fragment values are contiguous -> one 32B ds read
// per fragment.  B is stored [col][k].  Pitch 40 f16 (80B = 20 banks) makes
// fragment reads bank-conflict-free.  Bounds use clamp+select (no divergent
// loads); EXEC stays full through every WMMA.
// ---------------------------------------------------------------------------
static constexpr int AP = 40;                   // f16 pitch

struct TileRegs {
  float a[16];
  float b[8];
};

static __device__ __forceinline__
TileRegs load_tile(const float* __restrict__ A, long ars, long acs,
                   const float* __restrict__ B, long brs, long bcs,
                   int M, int N, int Kd, int gm0, int gn0, int k0, int t) {
  TileRegs r;
  // ---- A: rows gm0..+127, 32 k in perm order ----
  {
    int tr  = t >> 1;                           // 0..127
    int tk  = (t & 1) * 16;                     // perm chunk 0 or 16
    int row = gm0 + tr; if (row >= M) row = M - 1;   // clamp; OOB rows unused
    const float* base = A + (long)row * ars;
    if (acs == 1 && (k0 + 32) <= Kd && (ars & 3) == 0) {
      int o0 = tk ? 8 : 0;
      int o1 = tk ? 24 : 16;
      float4 f0 = *(const float4*)(base + k0 + o0);
      float4 f1 = *(const float4*)(base + k0 + o0 + 4);
      float4 f2 = *(const float4*)(base + k0 + o1);
      float4 f3 = *(const float4*)(base + k0 + o1 + 4);
      r.a[0]=f0.x;  r.a[1]=f0.y;  r.a[2]=f0.z;  r.a[3]=f0.w;
      r.a[4]=f1.x;  r.a[5]=f1.y;  r.a[6]=f1.z;  r.a[7]=f1.w;
      r.a[8]=f2.x;  r.a[9]=f2.y;  r.a[10]=f2.z; r.a[11]=f2.w;
      r.a[12]=f3.x; r.a[13]=f3.y; r.a[14]=f3.z; r.a[15]=f3.w;
    } else {
#pragma unroll
      for (int e = 0; e < 16; ++e) {
        int p = tk + e;                         // perm position 0..31
        int g = p >> 3;                         // swap groups 1 and 2
        int k = k0 + p + ((g == 1) ? 8 : (g == 2) ? -8 : 0);
        int kc = (k < Kd) ? k : (Kd - 1);
        float v = base[(long)kc * acs];
        r.a[e] = (k < Kd) ? v : 0.0f;
      }
    }
  }
  // ---- B: k = k0+kk, cols gn0+jc..+7 ----
  {
    int kk = t >> 3;                            // 0..31
    int jc = (t & 7) * 8;                       // 0,8,...,56
    if (bcs == 1 && (k0 + 32) <= Kd && (brs & 3) == 0 && (gn0 + 64) <= N) {
      const float* bp = B + (long)(k0 + kk) * brs + gn0 + jc;
      float4 f0 = *(const float4*)(bp);
      float4 f1 = *(const float4*)(bp + 4);
      r.b[0]=f0.x; r.b[1]=f0.y; r.b[2]=f0.z; r.b[3]=f0.w;
      r.b[4]=f1.x; r.b[5]=f1.y; r.b[6]=f1.z; r.b[7]=f1.w;
    } else {
      int k  = k0 + kk;
      int kc = (k < Kd) ? k : (Kd - 1);
      bool kok = (k < Kd);
      const float* bp = B + (long)kc * brs;
#pragma unroll
      for (int e = 0; e < 8; ++e) {
        int col = gn0 + jc + e;
        int cc  = (col < N) ? col : (N - 1);
        float v = bp[(long)cc * bcs];
        r.b[e] = (kok && col < N) ? v : 0.0f;
      }
    }
  }
  return r;
}

static __device__ __forceinline__
void store_tile(const TileRegs& r, _Float16* __restrict__ Ash,
                _Float16* __restrict__ Bsh, int t) {
  {
    int tr = t >> 1;
    int tk = (t & 1) * 16;
    v16h frag;
#pragma unroll
    for (int e = 0; e < 16; ++e) frag[e] = (_Float16)r.a[e];
    *(v16h*)&Ash[tr * AP + tk] = frag;
  }
  {
    int kk = t >> 3;
    int jc = (t & 7) * 8;
#pragma unroll
    for (int e = 0; e < 8; ++e)
      Bsh[(jc + e) * AP + kk + (((kk >> 3) == 1) ? 8 : ((kk >> 3) == 2) ? -8 : 0)] = (_Float16)r.b[e];
  }
}

__global__ __launch_bounds__(256)
void k_gemm_wmma(const float* __restrict__ A, long ars, long acs,
                 const float* __restrict__ B, long brs, long bcs,
                 float* __restrict__ C, long crs,
                 int M, int N, int Kd,
                 const float* __restrict__ bias, int relu) {
  __shared__ alignas(16) _Float16 Ash[2][128 * AP];
  __shared__ alignas(16) _Float16 Bsh[2][64 * AP];

  int t    = threadIdx.x;
  int lane = t & 31;
  int w    = t >> 5;
  int wm   = w >> 1;
  int wn   = w & 1;
  int gm0  = blockIdx.y * 128;
  int gn0  = blockIdx.x * 64;

  v8f acc00 = {}, acc01 = {}, acc10 = {}, acc11 = {};

  int ar0 = (wm * 32 + (lane & 15)) * AP + (lane >> 4) * 16;
  int bc0 = (wn * 32 + (lane & 15)) * AP + (lane >> 4) * 16;

  int nk = (Kd + 31) >> 5;
  TileRegs regs = load_tile(A, ars, acs, B, brs, bcs, M, N, Kd,
                            gm0, gn0, 0, t);
  for (int i = 0; i < nk; ++i) {
    int buf = i & 1;
    store_tile(regs, Ash[buf], Bsh[buf], t);
    __syncthreads();
    if (i + 1 < nk)                              // prefetch next tile while
      regs = load_tile(A, ars, acs, B, brs, bcs, // WMMAs below execute
                       M, N, Kd, gm0, gn0, (i + 1) * 32, t);
    v16h a0 = *(const v16h*)&Ash[buf][ar0];
    v16h a1 = *(const v16h*)&Ash[buf][ar0 + 16 * AP];
    v16h b0 = *(const v16h*)&Bsh[buf][bc0];
    v16h b1 = *(const v16h*)&Bsh[buf][bc0 + 16 * AP];
    acc00 = __builtin_amdgcn_wmma_f32_16x16x32_f16(false, a0, false, b0,
                                                   (short)0, acc00, false, false);
    acc01 = __builtin_amdgcn_wmma_f32_16x16x32_f16(false, a0, false, b1,
                                                   (short)0, acc01, false, false);
    acc10 = __builtin_amdgcn_wmma_f32_16x16x32_f16(false, a1, false, b0,
                                                   (short)0, acc10, false, false);
    acc11 = __builtin_amdgcn_wmma_f32_16x16x32_f16(false, a1, false, b1,
                                                   (short)0, acc11, false, false);
  }

  // ---- store C fragments (guarded) ----
  int gm = gm0 + wm * 32;
  int gn = gn0 + wn * 32;
  const v8f* accs[4] = {&acc00, &acc01, &acc10, &acc11};
#pragma unroll
  for (int f = 0; f < 4; ++f) {
    int row0  = gm + (f >> 1) * 16;
    int col   = gn + (f & 1) * 16 + (lane & 15);
    int rbase = row0 + ((lane >> 4) << 3);
    if (col < N) {
      float bv = bias ? bias[col] : 0.0f;
      const v8f& acc = *accs[f];
#pragma unroll
      for (int r = 0; r < 8; ++r) {
        int row = rbase + r;
        if (row < M) {
          float v = acc[r] + bv;
          if (relu) v = fmaxf(v, 0.0f);
          C[(long)row * crs + col] = v;
        }
      }
    }
  }
}

// ---------------------------------------------------------------------------
// Utility kernels
// ---------------------------------------------------------------------------
__global__ void k_zero1d(float* __restrict__ p, long n) {
  long t = (long)blockIdx.x * blockDim.x + threadIdx.x;
  long s = (long)gridDim.x * blockDim.x;
  for (; t < n; t += s) p[t] = 0.0f;
}

__global__ void k_zero2d(float* __restrict__ p, int cols, long stride) {
  long base = (long)blockIdx.x * stride;
  for (int c = threadIdx.x; c < cols; c += blockDim.x) p[base + c] = 0.0f;
}

// dst[didx[e], :] += src[sidx[e], :]   (one block per edge, f32 atomics)
__global__ void k_scatter(const float* __restrict__ src, long ss,
                          const int* __restrict__ sidx,
                          const int* __restrict__ didx,
                          int nf, float* __restrict__ dst, long ds) {
  int e = blockIdx.x;
  const float* sp = src + (long)sidx[e] * ss;
  float*       dp = dst + (long)didx[e] * ds;
  for (int f = threadIdx.x; f < nf; f += blockDim.x)
    atomicAdd(&dp[f], sp[f]);
}

// ---------------------------------------------------------------------------
// Farthest point sampling: one workgroup, positions + running min-dist in LDS
// (n*4 floats + 8KB reduction scratch; <=139KB << 320KB WGP LDS)
// ---------------------------------------------------------------------------
__global__ void k_fps(const float* __restrict__ pos, int ps,
                      int n, int ns, float* __restrict__ cent) {
  extern __shared__ float smem[];
  float* px   = smem;
  float* py   = px + n;
  float* pz   = py + n;
  float* dist = pz + n;
  float* rv   = dist + n;
  int*   ri   = (int*)(rv + blockDim.x);
  int tid = threadIdx.x;

  for (int i = tid; i < n; i += blockDim.x) {
    px[i] = pos[(long)i * ps + 0];
    py[i] = pos[(long)i * ps + 1];
    pz[i] = pos[(long)i * ps + 2];
    dist[i] = INFINITY;
  }
  __syncthreads();
  if (tid == 0) { cent[0] = px[0]; cent[1] = py[0]; cent[2] = pz[0]; }
  int cur = 0;
  for (int s = 1; s < ns; ++s) {
    float cx = px[cur], cy = py[cur], cz = pz[cur];
    float bv = -1.0f; int bi = 0x7fffffff;
    for (int i = tid; i < n; i += blockDim.x) {
      float dx = px[i] - cx, dy = py[i] - cy, dz = pz[i] - cz;
      float d  = dx * dx + dy * dy + dz * dz;
      float nd = fminf(dist[i], d);
      dist[i] = nd;
      if (nd > bv || (nd == bv && i < bi)) { bv = nd; bi = i; }
    }
    rv[tid] = bv; ri[tid] = bi;
    __syncthreads();
    for (int off = blockDim.x >> 1; off > 0; off >>= 1) {
      if (tid < off) {
        float ov = rv[tid + off]; int oi = ri[tid + off];
        if (ov > rv[tid] || (ov == rv[tid] && oi < ri[tid])) {
          rv[tid] = ov; ri[tid] = oi;
        }
      }
      __syncthreads();
    }
    cur = ri[0];
    if (tid == 0) {
      cent[(long)s * 3 + 0] = px[cur];
      cent[(long)s * 3 + 1] = py[cur];
      cent[(long)s * 3 + 2] = pz[cur];
    }
    __syncthreads();
  }
}

// ---------------------------------------------------------------------------
// s = softmax((d2 < R^2), axis=1): row has two values e/Z and 1/Z,
// Z = cnt*e + (m-cnt).  One workgroup per row.
// ---------------------------------------------------------------------------
__global__ void k_smat(const float* __restrict__ pos, int ps,
                       const float* __restrict__ cent,
                       int m, float* __restrict__ s) {
  __shared__ int red[256];
  int i = blockIdx.x;
  float x = pos[(long)i * ps + 0];
  float y = pos[(long)i * ps + 1];
  float z = pos[(long)i * ps + 2];
  int local = 0;
  for (int j = threadIdx.x; j < m; j += blockDim.x) {
    float dx = x - cent[j * 3 + 0];
    float dy = y - cent[j * 3 + 1];
    float dz = z - cent[j * 3 + 2];
    if (dx * dx + dy * dy + dz * dz < RADIUS2) local++;
  }
  red[threadIdx.x] = local;
  __syncthreads();
  for (int off = blockDim.x >> 1; off > 0; off >>= 1) {
    if (threadIdx.x < off) red[threadIdx.x] += red[threadIdx.x + off];
    __syncthreads();
  }
  int cnt = red[0];
  float Z    = (float)cnt * E_CONST + (float)(m - cnt);
  float sIn  = E_CONST / Z;
  float sOut = 1.0f / Z;
  for (int j = threadIdx.x; j < m; j += blockDim.x) {
    float dx = x - cent[j * 3 + 0];
    float dy = y - cent[j * 3 + 1];
    float dz = z - cent[j * 3 + 2];
    bool in = (dx * dx + dy * dy + dz * dz < RADIUS2);
    s[(long)i * m + j] = in ? sIn : sOut;
  }
}

__global__ void k_sym(float* __restrict__ a, int m) {
  long t = (long)blockIdx.x * blockDim.x + threadIdx.x;
  long total = (long)m * m;
  if (t >= total) return;
  int i = (int)(t / m), j = (int)(t % m);
  if (i > j) return;
  float v = 0.5f * (a[(long)i * m + j] + a[(long)j * m + i]);
  a[(long)i * m + j] = v;
  a[(long)j * m + i] = v;
}

// top-K smallest per row (stable tie -> lower index), like argsort()[:, :K]
__global__ void k_topk(const float* __restrict__ a, int m, int* __restrict__ idx) {
  __shared__ float row[2048];
  __shared__ float rv[256];
  __shared__ int   ri[256];
  int r = blockIdx.x;
  for (int j = threadIdx.x; j < m; j += blockDim.x) row[j] = a[(long)r * m + j];
  __syncthreads();
  for (int k = 0; k < TOPK; ++k) {
    float bv = INFINITY; int bi = m;
    for (int j = threadIdx.x; j < m; j += blockDim.x) {
      float v = row[j];
      if (v < bv || (v == bv && j < bi)) { bv = v; bi = j; }
    }
    rv[threadIdx.x] = bv; ri[threadIdx.x] = bi;
    __syncthreads();
    for (int off = blockDim.x >> 1; off > 0; off >>= 1) {
      if (threadIdx.x < off) {
        float ov = rv[threadIdx.x + off]; int oi = ri[threadIdx.x + off];
        if (ov < rv[threadIdx.x] ||
            (ov == rv[threadIdx.x] && oi < ri[threadIdx.x])) {
          rv[threadIdx.x] = ov; ri[threadIdx.x] = oi;
        }
      }
      __syncthreads();
    }
    if (threadIdx.x == 0) {
      idx[(long)r * TOPK + k] = ri[0];
      row[ri[0]] = INFINITY;           // exclude from next selection
    }
    __syncthreads();
  }
}

__global__ void k_edges(const float* __restrict__ a, const int* __restrict__ idx,
                        int m, int* __restrict__ snd, int* __restrict__ rcv,
                        float* __restrict__ ev) {
  int t = blockIdx.x * blockDim.x + threadIdx.x;
  if (t >= m * TOPK) return;
  int r = t / TOPK, k = t % TOPK;
  int s = idx[r * TOPK + 0];
  int c = idx[r * TOPK + k];
  snd[t] = s; rcv[t] = c;
  ev[t] = a[(long)s * m + c];
}

__global__ void k_finalize(const float* __restrict__ h, int hn,
                           const float* __restrict__ ev,
                           const int* __restrict__ snd,
                           const int* __restrict__ rcv,
                           int ne, float* __restrict__ out) {
  int total = hn + 3 * ne;
  for (int t = blockIdx.x * blockDim.x + threadIdx.x; t < total;
       t += gridDim.x * blockDim.x) {
    if (t < hn)                out[t] = h[t];
    else if (t < hn + ne)      out[t] = ev[t - hn];
    else if (t < hn + 2 * ne)  out[t] = (float)snd[t - hn - ne];
    else                       out[t] = (float)rcv[t - hn - 2 * ne];
  }
}

// ---------------------------------------------------------------------------
// Host-side orchestration (all launches on `stream`; no allocation/sync)
// ---------------------------------------------------------------------------
static void run_gnn(const float* h_in, long in_stride, int in_dim, int n,
                    const float* W_in, const float* b_in,
                    const float* W_layers, const float* b_layers,
                    const int* snd, const int* rcv, int E,
                    float* hcatA, float* hcatB, float* z, hipStream_t stream) {
  dim3 g((DH + 63) / 64, (n + 127) / 128);
  // encode: hcatA[:, :64] = relu(h_in @ W_in + b_in)   (row stride 128)
  k_gemm_wmma<<<g, 256, 0, stream>>>(h_in, in_stride, 1, W_in, DH, 1,
                                     hcatA, 128, n, DH, in_dim, b_in, 1);
  float* cur = hcatA; float* nxt = hcatB;
  for (int l = 0; l < NL; ++l) {
    k_zero2d<<<n, 64, 0, stream>>>(cur + DH, DH, 128);           // m-part := 0
    k_scatter<<<E, 64, 0, stream>>>(cur, 128, snd, rcv, DH, cur + DH, 128);
    float* dst = (l == NL - 1) ? z : nxt;
    long crs   = (l == NL - 1) ? DH : 128;
    k_gemm_wmma<<<g, 256, 0, stream>>>(cur, 128, 1,
                                       W_layers + (long)l * 2 * DH * DH, DH, 1,
                                       dst, crs, n, DH, 2 * DH,
                                       b_layers + (long)l * DH, 1);
    float* t = cur; cur = nxt; nxt = t;
  }
}

extern "C" void kernel_launch(void* const* d_in, const int* in_sizes, int n_in,
                              void* d_out, int out_size, void* d_ws, size_t ws_size,
                              hipStream_t stream) {
  (void)in_sizes; (void)n_in; (void)out_size; (void)ws_size;
  const float* nodes   = (const float*)d_in[0];
  const float* W0_in   = (const float*)d_in[1];
  const float* b0_in   = (const float*)d_in[2];
  const float* W0_lay  = (const float*)d_in[3];
  const float* b0_lay  = (const float*)d_in[4];
  const float* W1_in   = (const float*)d_in[5];
  const float* b1_in   = (const float*)d_in[6];
  const float* W1_lay  = (const float*)d_in[7];
  const float* b1_lay  = (const float*)d_in[8];
  const int*   senders   = (const int*)d_in[9];
  const int*   receivers = (const int*)d_in[10];
  float* out = (float*)d_out;

  // ---- carve workspace (s0/As0 = 64MB each: both stay L2-resident) ----
  char* ws = (char*)d_ws;
  size_t off = 0;
  auto alloc = [&](size_t bytes) -> void* {
    void* p = ws + off;
    off += (bytes + 255) & ~(size_t)255;
    return p;
  };
  float* hcatA = (float*)alloc((size_t)N0 * 128 * 4);
  float* hcatB = (float*)alloc((size_t)N0 * 128 * 4);
  float* z0    = (float*)alloc((size_t)N0 * DH * 4);
  float* cent0 = (float*)alloc((size_t)M0 * 3 * 4);
  float* s0    = (float*)alloc((size_t)N0 * M0 * 4);
  float* As0   = (float*)alloc((size_t)N0 * M0 * 4);
  float* a0    = (float*)alloc((size_t)M0 * M0 * 4);
  float* h1    = (float*)alloc((size_t)M0 * DH * 4);
  int*   idx0  = (int*)  alloc((size_t)M0 * TOPK * 4);
  int*   snd1  = (int*)  alloc((size_t)M0 * TOPK * 4);
  int*   rcv1  = (int*)  alloc((size_t)M0 * TOPK * 4);
  float* ev1   = (float*)alloc((size_t)M0 * TOPK * 4);
  float* z1    = (float*)alloc((size_t)M0 * DH * 4);
  float* cent1 = (float*)alloc((size_t)M1 * 3 * 4);
  float* s1    = (float*)alloc((size_t)M0 * M1 * 4);
  float* As1   = (float*)alloc((size_t)M0 * M1 * 4);
  float* a1    = (float*)alloc((size_t)M1 * M1 * 4);
  float* h2    = (float*)alloc((size_t)M1 * DH * 4);
  int*   idx1  = (int*)  alloc((size_t)M1 * TOPK * 4);
  int*   snd2  = (int*)  alloc((size_t)M1 * TOPK * 4);
  int*   rcv2  = (int*)  alloc((size_t)M1 * TOPK * 4);
  float* ev2   = (float*)alloc((size_t)M1 * TOPK * 4);

  // ======================= Stage 0 (n=8192 -> m=2048) =======================
  run_gnn(nodes, DIN, DIN, N0, W0_in, b0_in, W0_lay, b0_lay,
          senders, receivers, E0, hcatA, hcatB, z0, stream);

  size_t fps_sh0 = (size_t)N0 * 4 * 4 + 1024 * 8;
  k_fps<<<1, 1024, fps_sh0, stream>>>(nodes, DIN, N0, M0, cent0);
  k_smat<<<N0, 256, 0, stream>>>(nodes, DIN, cent0, M0, s0);

  k_zero1d<<<4096, 256, 0, stream>>>(As0, (long)N0 * M0);
  // As0 = segment_sum(s0[receivers]) by senders
  k_scatter<<<E0, 256, 0, stream>>>(s0, M0, receivers, senders, M0, As0, M0);

  // a0 = s0^T @ As0   (M=2048, N=2048, K=8192) — dominant WMMA GEMM
  {
    dim3 g((M0 + 63) / 64, (M0 + 127) / 128);
    k_gemm_wmma<<<g, 256, 0, stream>>>(s0, 1, M0, As0, M0, 1,
                                       a0, M0, M0, M0, N0, nullptr, 0);
  }
  k_sym<<<(int)(((long)M0 * M0 + 255) / 256), 256, 0, stream>>>(a0, M0);

  // h1 = s0^T @ z0   (M=2048, N=64, K=8192)
  {
    dim3 g((DH + 63) / 64, (M0 + 127) / 128);
    k_gemm_wmma<<<g, 256, 0, stream>>>(s0, 1, M0, z0, DH, 1,
                                       h1, DH, M0, DH, N0, nullptr, 0);
  }

  k_topk<<<M0, 256, 0, stream>>>(a0, M0, idx0);
  k_edges<<<(M0 * TOPK + 255) / 256, 256, 0, stream>>>(a0, idx0, M0,
                                                       snd1, rcv1, ev1);

  // ======================= Stage 1 (n=2048 -> m=512) ========================
  run_gnn(h1, DH, DH, M0, W1_in, b1_in, W1_lay, b1_lay,
          snd1, rcv1, M0 * TOPK, hcatA, hcatB, z1, stream);

  size_t fps_sh1 = (size_t)M0 * 4 * 4 + 1024 * 8;
  k_fps<<<1, 1024, fps_sh1, stream>>>(cent0, 3, M0, M1, cent1);
  k_smat<<<M0, 256, 0, stream>>>(cent0, 3, cent1, M1, s1);

  k_zero1d<<<1024, 256, 0, stream>>>(As1, (long)M0 * M1);
  k_scatter<<<M0 * TOPK, 256, 0, stream>>>(s1, M1, rcv1, snd1, M1, As1, M1);

  {
    dim3 g((M1 + 63) / 64, (M1 + 127) / 128);
    k_gemm_wmma<<<g, 256, 0, stream>>>(s1, 1, M1, As1, M1, 1,
                                       a1, M1, M1, M1, M0, nullptr, 0);
  }
  k_sym<<<(int)(((long)M1 * M1 + 255) / 256), 256, 0, stream>>>(a1, M1);

  {
    dim3 g((DH + 63) / 64, (M1 + 127) / 128);
    k_gemm_wmma<<<g, 256, 0, stream>>>(s1, 1, M1, z1, DH, 1,
                                       h2, DH, M1, DH, M0, nullptr, 0);
  }

  k_topk<<<M1, 256, 0, stream>>>(a1, M1, idx1);
  k_edges<<<(M1 * TOPK + 255) / 256, 256, 0, stream>>>(a1, idx1, M1,
                                                       snd2, rcv2, ev2);

  // ===================== Assemble tuple output ==============================
  k_finalize<<<256, 256, 0, stream>>>(h2, M1 * DH, ev2, snd2, rcv2,
                                      M1 * TOPK, out);
}
